// Network_80187039416351
// MI455X (gfx1250) — compile-verified, matrix operand order checked by reading, not run on hardware
//
#include <hip/hip_runtime.h>
#include <hip/hip_bf16.h>

// ---------------------------------------------------------------------------
// PointNet++ grasp network forward for MI455X (gfx1250, wave32, WMMA).
// All dense/MLP layers (SA MLPs, FP MLPs, fc, and the big head m1 layers) run
// through a register-blocked WMMA GEMM: each wave computes a 16x(16*NT) strip,
// reusing one A fragment against NT B fragments per 32-K step -> per step:
// 2 A b128 loads + 2*NT B b128 loads + NT independent v_wmma ops.
// Weights are transposed+padded to f16 once per launch; activations are
// produced in f16 directly so the K-loop is branch-free and fully vectorized.
// ---------------------------------------------------------------------------

typedef __attribute__((ext_vector_type(16))) _Float16 v16h;
typedef __attribute__((ext_vector_type(8)))  _Float16 v8h;
typedef __attribute__((ext_vector_type(8)))  float    v8f;

#define BATCH 16
#define NSAMPLE 32
#define RV_CNT 100

static inline int ceil_div(int a, int b) { return (a + b - 1) / b; }
static inline int round32(int k) { return (k + 31) & ~31; }

// ---------------------------------------------------------------------------
// WMMA GEMM: out[m,n] = act( sum_k A[m,k]*B[k,n] + bias[n] )
// A:  f16 (M, lda) row-major, lda multiple of 32 (zero padded past true K).
// Bt: f16 (N, ldb) = B^T, ldb multiple of 32, zero padded. K = padded K.
// A-layout (16-bit A 16x32, ISA 7.12.2): lane L holds row (L&15); v16h elem j
// holds K = k0 + (j<8 ? 8*(L>>4)+j : 16+8*(L>>4)+(j-8)) -> two contiguous
// 8-half runs at +hi*8 and +16+hi*8. B mirrors it with N in place of M.
// D-layout: VGPR r, lane L -> M = r + 8*(L>>4), N = (L&15).
// out: half row-major (rs=ld) if out_half, else float with strides (rs, cs).
// ---------------------------------------------------------------------------
template <int NT>
__global__ __launch_bounds__(128)
void wmma_gemm_nt(const _Float16* __restrict__ A, int lda,
                  const _Float16* __restrict__ Bt, int ldb,
                  const float* __restrict__ bias, int N,
                  void* __restrict__ outv, long rs, long cs, int out_half,
                  int K, int act, int ntiles)
{
    int wave = threadIdx.x >> 5;
    int lane = threadIdx.x & 31;
    int tile = blockIdx.x * (blockDim.x >> 5) + wave;
    if (tile >= ntiles) return;

    int ntn = N / (16 * NT);            // strips per output row of tiles
    int tm  = tile / ntn;
    int tn0 = (tile - tm * ntn) * NT;
    int r16 = lane & 15;
    int hi  = lane >> 4;

    const _Float16* arow = A + (size_t)(tm * 16 + r16) * lda + hi * 8;
    const _Float16* brow[NT];
#pragma unroll
    for (int t = 0; t < NT; ++t)
        brow[t] = Bt + (size_t)((tn0 + t) * 16 + r16) * ldb + hi * 8;

    v8f zero = {};
    v8f acc[NT];
#pragma unroll
    for (int t = 0; t < NT; ++t) acc[t] = zero;

    for (int k0 = 0; k0 < K; k0 += 32) {
        if (k0 + 32 < K) {
            __builtin_prefetch(arow + k0 + 32, 0, 1);
            __builtin_prefetch(brow[0] + k0 + 32, 0, 1);
        }
        union { v16h v; v8h h[2]; } ua;
        ua.h[0] = *(const v8h*)(arow + k0);
        ua.h[1] = *(const v8h*)(arow + k0 + 16);
        union { v16h v; v8h h[2]; } ub[NT];
#pragma unroll
        for (int t = 0; t < NT; ++t) {
            ub[t].h[0] = *(const v8h*)(brow[t] + k0);
            ub[t].h[1] = *(const v8h*)(brow[t] + k0 + 16);
        }
#pragma unroll
        for (int t = 0; t < NT; ++t)
            acc[t] = __builtin_amdgcn_wmma_f32_16x16x32_f16(
                false, ua.v, false, ub[t].v, (short)0, acc[t], false, false);
    }

#pragma unroll
    for (int t = 0; t < NT; ++t) {
        int ncol = (tn0 + t) * 16 + r16;
        float bn = bias[ncol];
#pragma unroll
        for (int r = 0; r < 8; ++r) {
            float v = acc[t][r] + bn;
            if (act == 1)      v = fmaxf(v, 0.0f);
            else if (act == 2) v = (v > 0.0f) ? v : 0.01f * v;
            size_t m = (size_t)(tm * 16 + r + 8 * hi);
            if (out_half)
                ((_Float16*)outv)[m * (size_t)rs + (size_t)ncol] = (_Float16)v;
            else
                ((float*)outv)[m * (size_t)rs + (size_t)ncol * (size_t)cs] = v;
        }
    }
}

// W (K,N) f32 row-major -> Wt (N,Kp) f16 row-major, zero padded.
__global__ void prep_weight(const float* __restrict__ W,
                            _Float16* __restrict__ Wt,
                            int K, int N, int Kp)
{
    int gid = blockIdx.x * blockDim.x + threadIdx.x;
    if (gid >= N * Kp) return;
    int n = gid / Kp;
    int k = gid - n * Kp;
    Wt[gid] = (k < K) ? (_Float16)W[(size_t)k * N + n] : (_Float16)0.0f;
}

// ---------------------------------------------------------------------------
// Farthest point sampling: one block per batch. idx[0]=0; idx[s] = argmax of
// dmin after update with pts[idx[s-1]] (first index on ties, like jnp.argmax).
// ---------------------------------------------------------------------------
__global__ void fps_kernel(const float* __restrict__ xyz, int n, int npoint,
                           int* __restrict__ idx)
{
    __shared__ float dmin[4096];
    __shared__ float rv[256];
    __shared__ int   ri[256];
    __shared__ int   s_sel;

    int b = blockIdx.x;
    int t = threadIdx.x;
    const float* pts = xyz + (size_t)b * n * 3;

    for (int i = t; i < n; i += blockDim.x) dmin[i] = 1e30f;
    if (t == 0) { s_sel = 0; idx[b * npoint] = 0; }
    __syncthreads();

    for (int s = 1; s < npoint; ++s) {
        int sel = s_sel;
        float px = pts[sel * 3 + 0];
        float py = pts[sel * 3 + 1];
        float pz = pts[sel * 3 + 2];
        float best = -1.0f;
        int   bi = 0x7fffffff;
        for (int i = t; i < n; i += blockDim.x) {
            float dx = pts[i * 3 + 0] - px;
            float dy = pts[i * 3 + 1] - py;
            float dz = pts[i * 3 + 2] - pz;
            float d = dx * dx + dy * dy + dz * dz;
            float dm = dmin[i];
            if (d < dm) { dm = d; dmin[i] = dm; }
            if (dm > best) { best = dm; bi = i; }
        }
        rv[t] = best; ri[t] = bi;
        __syncthreads();
        for (int off = blockDim.x / 2; off > 0; off >>= 1) {
            if (t < off) {
                if (rv[t + off] > rv[t] ||
                    (rv[t + off] == rv[t] && ri[t + off] < ri[t])) {
                    rv[t] = rv[t + off];
                    ri[t] = ri[t + off];
                }
            }
            __syncthreads();
        }
        if (t == 0) { s_sel = ri[0]; idx[b * npoint + s] = ri[0]; }
        __syncthreads();
    }
}

__global__ void gather_xyz(const float* __restrict__ xyz,
                           const int* __restrict__ idx,
                           float* __restrict__ out, int n, int np)
{
    int gid = blockIdx.x * blockDim.x + threadIdx.x;
    if (gid >= BATCH * np) return;
    int b = gid / np;
    int pi = idx[gid];
    const float* s = xyz + ((size_t)b * n + pi) * 3;
    out[(size_t)gid * 3 + 0] = s[0];
    out[(size_t)gid * 3 + 1] = s[1];
    out[(size_t)gid * 3 + 2] = s[2];
}

// First NSAMPLE source indices (ascending index order) within radius; pad with
// the first hit — matches sort(where(d2<=r2, i, n))[:32] + pad semantics.
__global__ void ball_query(const float* __restrict__ xyz,
                           const float* __restrict__ nxyz,
                           int n, int np, float r2, int* __restrict__ gi)
{
    int gid = blockIdx.x * blockDim.x + threadIdx.x;
    if (gid >= BATCH * np) return;
    int b = gid / np;
    const float* src = xyz + (size_t)b * n * 3;
    float qx = nxyz[(size_t)gid * 3 + 0];
    float qy = nxyz[(size_t)gid * 3 + 1];
    float qz = nxyz[(size_t)gid * 3 + 2];
    int* out = gi + (size_t)gid * NSAMPLE;
    int cnt = 0;
    for (int i = 0; i < n; ++i) {
        float dx = src[i * 3 + 0] - qx;
        float dy = src[i * 3 + 1] - qy;
        float dz = src[i * 3 + 2] - qz;
        if (dx * dx + dy * dy + dz * dz <= r2) {
            out[cnt++] = i;
            if (cnt == NSAMPLE) break;
        }
    }
    if (cnt == 0) { out[0] = 0; cnt = 1; }
    int p0 = out[0];
    for (int s = cnt; s < NSAMPLE; ++s) out[s] = p0;
}

// Grouped rows in f16: [rel_xyz(3) | feats(cf) | zero pad to cinp].
__global__ void group_points_h(const float* __restrict__ xyz,
                               const float* __restrict__ nxyz,
                               const float* __restrict__ feat,
                               const int* __restrict__ gi,
                               _Float16* __restrict__ out,
                               int np, int cf, int cinp)
{
    int cin = 3 + cf;
    int total = np * NSAMPLE * cinp;
    int gid = blockIdx.x * blockDim.x + threadIdx.x;
    if (gid >= total) return;
    int row = gid / cinp;
    int c = gid - row * cinp;
    int j = row >> 5;
    float v = 0.0f;
    if (c < cin) {
        int pi = gi[row];
        if (c < 3) v = xyz[pi * 3 + c] - nxyz[j * 3 + c];
        else       v = feat[(size_t)pi * cf + (c - 3)];
    }
    out[gid] = (_Float16)v;
}

__global__ void maxpool32(const float* __restrict__ in,
                          float* __restrict__ out, int np, int c)
{
    int gid = blockIdx.x * blockDim.x + threadIdx.x;
    if (gid >= np * c) return;
    int j = gid / c;
    int ch = gid - j * c;
    const float* p = in + ((size_t)j * NSAMPLE) * c + ch;
    float m = p[0];
    for (int s = 1; s < NSAMPLE; ++s) m = fmaxf(m, p[(size_t)s * c]);
    out[gid] = m;
}

// 3-NN interpolation + concat([interp(c2), f1(c1)]) -> f16 padded row (cwp).
__global__ void three_interp_h(const float* __restrict__ xyz1,
                               const float* __restrict__ xyz2,
                               const float* __restrict__ f1,
                               const float* __restrict__ f2,
                               _Float16* __restrict__ out,
                               int n1, int n2, int c1, int c2, int cwp)
{
    int j = blockIdx.x * blockDim.x + threadIdx.x;
    if (j >= n1) return;
    float qx = xyz1[j * 3 + 0], qy = xyz1[j * 3 + 1], qz = xyz1[j * 3 + 2];
    float d0 = 1e30f, d1 = 1e30f, d2v = 1e30f;
    int i0 = 0, i1 = 0, i2 = 0;
    for (int i = 0; i < n2; ++i) {
        float dx = xyz2[i * 3 + 0] - qx;
        float dy = xyz2[i * 3 + 1] - qy;
        float dz = xyz2[i * 3 + 2] - qz;
        float d = dx * dx + dy * dy + dz * dz;
        if (d < d0)       { d2v = d1; i2 = i1; d1 = d0; i1 = i0; d0 = d; i0 = i; }
        else if (d < d1)  { d2v = d1; i2 = i1; d1 = d;  i1 = i; }
        else if (d < d2v) { d2v = d;  i2 = i; }
    }
    float w0 = 1.0f / (d0 + 1e-8f);
    float w1 = 1.0f / (d1 + 1e-8f);
    float w2 = 1.0f / (d2v + 1e-8f);
    float wsum = w0 + w1 + w2;
    w0 /= wsum; w1 /= wsum; w2 /= wsum;
    _Float16* orow = out + (size_t)j * cwp;
    const float* p0 = f2 + (size_t)i0 * c2;
    const float* p1 = f2 + (size_t)i1 * c2;
    const float* p2 = f2 + (size_t)i2 * c2;
    for (int c = 0; c < c2; ++c)
        orow[c] = (_Float16)(w0 * p0[c] + w1 * p1[c] + w2 * p2[c]);
    const float* fr = f1 + (size_t)j * c1;
    for (int c = 0; c < c1; ++c) orow[c2 + c] = (_Float16)fr[c];
    for (int c = c1 + c2; c < cwp; ++c) orow[c] = (_Float16)0.0f;
}

// ---------------------------------------------------------------------------
// Heads: scalar dense for tiny-N layers, builders, bgs/bgdR, losses.
// act: 0 linear, 1 relu, 2 leaky(0.01), 3 sigmoid
// ---------------------------------------------------------------------------
__global__ void dense_small(const float* __restrict__ in,
                            const float* __restrict__ W,
                            const float* __restrict__ bias,
                            float* __restrict__ out,
                            int M, int K, int N, int act)
{
    int gid = blockIdx.x * blockDim.x + threadIdx.x;
    if (gid >= M * N) return;
    int m = gid / N;
    int n = gid - m * N;
    float s = bias[n];
    const float* ir = in + (size_t)m * K;
    for (int k = 0; k < K; ++k) s += ir[k] * W[(size_t)k * N + n];
    if (act == 1)      s = fmaxf(s, 0.0f);
    else if (act == 2) s = (s > 0.0f) ? s : 0.01f * s;
    else if (act == 3) s = 1.0f / (1.0f + expf(-s));
    out[gid] = s;
}

__global__ void extract_net(const float* __restrict__ whole,
                            float* __restrict__ net,
                            _Float16* __restrict__ net_h)
{
    int gid = blockIdx.x * blockDim.x + threadIdx.x;
    if (gid >= BATCH * 128) return;
    int b = gid >> 7;
    int c = gid & 127;
    float v = whole[(size_t)b * 128 * 4096 + (size_t)c * 4096]; // point 0
    net[gid] = v;
    net_h[gid] = (_Float16)v;
}

// [net(128) | dirs1(3) | dirs2(3) | gd(1) | pad] -> (16,160) f16
__global__ void build_crit1_h(const float* __restrict__ net,
                              const float* __restrict__ dirs1,
                              const float* __restrict__ dirs2,
                              const float* __restrict__ gd,
                              _Float16* __restrict__ out)
{
    int gid = blockIdx.x * blockDim.x + threadIdx.x;
    if (gid >= BATCH * 160) return;
    int b = gid / 160;
    int c = gid - b * 160;
    float v = 0.0f;
    if (c < 128)      v = net[b * 128 + c];
    else if (c < 131) v = dirs1[b * 3 + (c - 128)];
    else if (c < 134) v = dirs2[b * 3 + (c - 131)];
    else if (c < 135) v = gd[b];
    out[gid] = (_Float16)v;
}

// [net[r/100](128) | rvs[r](10) | pad] -> (1600,160) f16
__global__ void build_actor_in_h(const float* __restrict__ net,
                                 const float* __restrict__ rvs,
                                 _Float16* __restrict__ out)
{
    int gid = blockIdx.x * blockDim.x + threadIdx.x;
    if (gid >= BATCH * RV_CNT * 160) return;
    int r = gid / 160;
    int c = gid - r * 160;
    float v = 0.0f;
    if (c < 128)      v = net[(r / RV_CNT) * 128 + c];
    else if (c < 138) v = rvs[(size_t)r * 10 + (c - 128)];
    out[gid] = (_Float16)v;
}

// [net[r/100](128) | pred6[r](6) | gd[r%16](1) | pad] -> (1600,160) f16
__global__ void build_crit2_h(const float* __restrict__ net,
                              const float* __restrict__ pred6,
                              const float* __restrict__ gd,
                              _Float16* __restrict__ out)
{
    int gid = blockIdx.x * blockDim.x + threadIdx.x;
    if (gid >= BATCH * RV_CNT * 160) return;
    int r = gid / 160;
    int c = gid - r * 160;
    float v = 0.0f;
    if (c < 128)      v = net[(r / RV_CNT) * 128 + c];
    else if (c < 134) v = pred6[(size_t)r * 6 + (c - 128)];
    else if (c < 135) v = gd[r % BATCH];   // jnp.tile(gd,(RV_CNT,1))
    out[gid] = (_Float16)v;
}

__device__ inline void bgs_cols_dev(const float u[3], const float v[3], float R[9])
{
    float nu = sqrtf(u[0]*u[0] + u[1]*u[1] + u[2]*u[2]);
    nu = fmaxf(nu, 1e-12f);
    float b1x = u[0]/nu, b1y = u[1]/nu, b1z = u[2]/nu;
    float d = b1x*v[0] + b1y*v[1] + b1z*v[2];
    float tx = v[0] - d*b1x, ty = v[1] - d*b1y, tz = v[2] - d*b1z;
    float nt = sqrtf(tx*tx + ty*ty + tz*tz);
    nt = fmaxf(nt, 1e-12f);
    float b2x = tx/nt, b2y = ty/nt, b2z = tz/nt;
    float b3x = b1y*b2z - b1z*b2y;
    float b3y = b1z*b2x - b1x*b2z;
    float b3z = b1x*b2y - b1y*b2x;
    R[0]=b1x; R[1]=b2x; R[2]=b3x;
    R[3]=b1y; R[4]=b2y; R[5]=b3y;
    R[6]=b1z; R[7]=b2z; R[8]=b3z;
}

// o (M,6) -> pred6 = bgs(o.reshape(-1,3,2))[:,:,:2].reshape(-1,6)
__global__ void actor_bgs(const float* __restrict__ o,
                          float* __restrict__ pred6, int M)
{
    int r = blockIdx.x * blockDim.x + threadIdx.x;
    if (r >= M) return;
    const float* p = o + (size_t)r * 6;
    float a1[3] = { p[0], p[2], p[4] };   // column 0 of (3,2)
    float a2[3] = { p[1], p[3], p[5] };   // column 1
    float R[9];
    bgs_cols_dev(a1, a2, R);
    float* q = pred6 + (size_t)r * 6;
    q[0] = R[0]; q[1] = R[1];
    q[2] = R[3]; q[3] = R[4];
    q[4] = R[6]; q[5] = R[7];
}

// theta[r] = bgdR(bgs(to_cols(gt6)), bgs(to_cols(pred6)));
// to_cols(v6): col0 = v[0:3], col1 = v[3:6]
__global__ void theta_kernel(const float* __restrict__ pred6,
                             const float* __restrict__ dirs1,
                             const float* __restrict__ dirs2,
                             float* __restrict__ theta, int M)
{
    int r = blockIdx.x * blockDim.x + threadIdx.x;
    if (r >= M) return;
    int b = r / RV_CNT;
    float u0[3] = { dirs1[b*3+0], dirs1[b*3+1], dirs1[b*3+2] };
    float u1[3] = { dirs2[b*3+0], dirs2[b*3+1], dirs2[b*3+2] };
    float Rg[9];
    bgs_cols_dev(u0, u1, Rg);
    const float* p = pred6 + (size_t)r * 6;
    float v0[3] = { p[0], p[1], p[2] };
    float v1[3] = { p[3], p[4], p[5] };
    float Rp[9];
    bgs_cols_dev(v0, v1, Rp);
    float Rt = 0.0f;
    for (int i = 0; i < 9; ++i) Rt += Rg[i] * Rp[i];
    float c = 0.5f * (Rt - 1.0f);
    c = fminf(fmaxf(c, -1.0f + 1e-6f), 1.0f - 1e-6f);
    theta[r] = acosf(c);
}

__global__ void min100_k(const float* __restrict__ t, float* __restrict__ out)
{
    int b = threadIdx.x;
    if (b >= BATCH) return;
    float m = t[b * RV_CNT];
    for (int j = 1; j < RV_CNT; ++j) m = fminf(m, t[b * RV_CNT + j]);
    out[b] = m;
}

__global__ void mean100_k(const float* __restrict__ t, float* __restrict__ out)
{
    int b = threadIdx.x;
    if (b >= BATCH) return;
    float s = 0.0f;
    for (int j = 0; j < RV_CNT; ++j) s += t[b * RV_CNT + j];
    out[b] = s * (1.0f / RV_CNT);
}

__global__ void width_loss_k(const float* __restrict__ gd,
                             const float* __restrict__ gtw,
                             float* __restrict__ out)
{
    int b = threadIdx.x;
    if (b >= BATCH) return;
    float d = gd[b] - gtw[b];
    out[b] = d * d;
}

__global__ void critic_loss_k(const float* __restrict__ logits,
                              const float* __restrict__ gtr,
                              float* __restrict__ out)
{
    int b = threadIdx.x;
    if (b >= BATCH) return;
    float l = logits[b];
    float x = -l;                                   // softplus(-logits)
    float sp = fmaxf(x, 0.0f) + log1pf(expf(-fabsf(x)));
    out[b] = (1.0f - gtr[b]) * l + sp;
}

__global__ void as_loss_k(const float* __restrict__ pas,
                          const float* __restrict__ avg,
                          float* __restrict__ out)
{
    int b = threadIdx.x;
    if (b >= BATCH) return;
    float d = pas[b] - avg[b];
    out[b] = d * d;
}

// ---------------------------------------------------------------------------
// Host orchestration
// ---------------------------------------------------------------------------
static void gemm(const _Float16* A, int lda, const _Float16* Bt, int ldb,
                 const float* bias, int N, void* out, long rs, long cs,
                 int out_half, int M, int Kp, int act, hipStream_t stream)
{
    if ((N & 63) == 0) {
        int ntiles = (M / 16) * (N / 64);
        wmma_gemm_nt<4><<<ceil_div(ntiles, 4), 128, 0, stream>>>(
            A, lda, Bt, ldb, bias, N, out, rs, cs, out_half, Kp, act, ntiles);
    } else {                                  // N == 32
        int ntiles = (M / 16) * (N / 32);
        wmma_gemm_nt<2><<<ceil_div(ntiles, 4), 128, 0, stream>>>(
            A, lda, Bt, ldb, bias, N, out, rs, cs, out_half, Kp, act, ntiles);
    }
}

extern "C" void kernel_launch(void* const* d_in, const int* in_sizes, int n_in,
                              void* d_out, int out_size, void* d_ws, size_t ws_size,
                              hipStream_t stream)
{
    (void)in_sizes; (void)n_in; (void)out_size; (void)ws_size;

    const float* pcs    = (const float*)d_in[0];
    const float* dirs1  = (const float*)d_in[1];
    const float* dirs2  = (const float*)d_in[2];
    const float* gtw    = (const float*)d_in[3];
    const float* gtr    = (const float*)d_in[4];
    const float* rvs    = (const float*)d_in[5];

    // params pytree order: actor, ascore, critic, fc, fp, gdepth, sa
    const float* actW1 = (const float*)d_in[6];
    const float* actB1 = (const float*)d_in[7];
    const float* actW2 = (const float*)d_in[8];
    const float* actB2 = (const float*)d_in[9];
    const float* asW1  = (const float*)d_in[10];
    const float* asB1  = (const float*)d_in[11];
    const float* asW2  = (const float*)d_in[12];
    const float* asB2  = (const float*)d_in[13];
    const float* crW1  = (const float*)d_in[14];
    const float* crB1  = (const float*)d_in[15];
    const float* crW2  = (const float*)d_in[16];
    const float* crB2  = (const float*)d_in[17];
    const float* fcW   = (const float*)d_in[18];
    const float* fcB   = (const float*)d_in[19];
    const float* fpW[4][3] = {}; const float* fpB[4][3] = {};
    {
        int base = 20;
        const int nl[4] = {3, 2, 2, 2};
        for (int i = 0; i < 4; ++i) {
            for (int l = 0; l < nl[i]; ++l) {
                fpW[i][l] = (const float*)d_in[base + l * 2];
                fpB[i][l] = (const float*)d_in[base + l * 2 + 1];
            }
            base += nl[i] * 2;
        }
    }
    const float* gdW1 = (const float*)d_in[38];
    const float* gdB1 = (const float*)d_in[39];
    const float* gdW2 = (const float*)d_in[40];
    const float* gdB2 = (const float*)d_in[41];
    const float* saW[4][3]; const float* saB[4][3];
    for (int i = 0; i < 4; ++i)
        for (int l = 0; l < 3; ++l) {
            saW[i][l] = (const float*)d_in[42 + i * 6 + l * 2];
            saB[i][l] = (const float*)d_in[42 + i * 6 + l * 2 + 1];
        }

    // ---- workspace bump allocator (bytes, 64B aligned) ----
    char* wsb = (char*)d_ws;
    size_t off = 0;
    auto allocb = [&](size_t bytes) {
        void* p = wsb + off;
        off += (bytes + 63) & ~(size_t)63;
        return p;
    };
    auto allocf = [&](size_t n) { return (float*)allocb(n * 4); };
    auto alloch = [&](size_t n) { return (_Float16*)allocb(n * 2); };

    float* xyz1  = allocf((size_t)BATCH * 1024 * 3);
    float* xyz2  = allocf((size_t)BATCH * 256 * 3);
    float* xyz3  = allocf((size_t)BATCH * 64 * 3);
    float* xyz4  = allocf((size_t)BATCH * 16 * 3);
    float* f1    = allocf((size_t)BATCH * 1024 * 64);
    float* f2    = allocf((size_t)BATCH * 256 * 128);
    float* f3    = allocf((size_t)BATCH * 64 * 256);
    float* f4    = allocf((size_t)BATCH * 16 * 512);
    float* nf3   = allocf((size_t)BATCH * 64 * 256);
    float* nf2   = allocf((size_t)BATCH * 256 * 256);
    float* nf1   = allocf((size_t)BATCH * 1024 * 128);
    float* bufF  = allocf((size_t)2097152);          // max 32768x64 f32
    _Float16* bufA_h = alloch((size_t)2097152);      // max 32768x32 / 4096x160
    _Float16* bufB_h = alloch((size_t)2097152);
    _Float16* f0fin  = alloch((size_t)BATCH * 4096 * 128);
    int*   fpsidx = (int*)allocb((size_t)BATCH * 1024 * 4);
    int*   gidx   = (int*)allocb((size_t)BATCH * 1024 * NSAMPLE * 4);
    float* net    = allocf((size_t)BATCH * 128);
    _Float16* net_h = alloch((size_t)BATCH * 128);
    _Float16* in_h  = alloch((size_t)BATCH * RV_CNT * 160);
    float* h1    = allocf((size_t)BATCH * RV_CNT * 128);
    float* gdv   = allocf(BATCH);
    float* o6    = allocf((size_t)BATCH * RV_CNT * 6);
    float* pred6 = allocf((size_t)BATCH * RV_CNT * 6);
    float* theta = allocf((size_t)BATCH * RV_CNT);
    float* prop  = allocf((size_t)BATCH * RV_CNT);
    float* pas   = allocf(BATCH);
    float* avg   = allocf(BATCH);

    float* outp  = (float*)d_out;
    float* critic_loss = outp + 0;
    float* actor_cov   = outp + 16;
    float* as_loss     = outp + 32;
    float* width_loss  = outp + 48;
    float* logits      = outp + 64;
    float* whole       = outp + 80;   // (16,128,4096)

    // ---- transpose + f16-convert + K-pad all GEMM weights ----
    const int SA_K[4][3] = {{6,32,32},{67,64,64},{131,128,128},{259,256,256}};
    const int SA_N[4][3] = {{32,32,64},{64,64,128},{128,128,256},{256,256,512}};
    const int FP_K[4][3] = {{131,128,128},{320,256,0},{384,256,0},{768,256,0}};
    const int FP_N[4][3] = {{128,128,128},{256,128,0},{256,256,0},{256,256,0}};
    const int FP_NL[4] = {3, 2, 2, 2};

    auto prep = [&](const float* W, int K, int N) {
        int Kp = round32(K);
        _Float16* Wt = alloch((size_t)N * Kp);
        prep_weight<<<ceil_div(N * Kp, 256), 256, 0, stream>>>(W, Wt, K, N, Kp);
        return Wt;
    };
    _Float16* saWt[4][3]; _Float16* fpWt[4][3] = {};
    for (int i = 0; i < 4; ++i)
        for (int l = 0; l < 3; ++l)
            saWt[i][l] = prep(saW[i][l], SA_K[i][l], SA_N[i][l]);
    for (int i = 0; i < 4; ++i)
        for (int l = 0; l < FP_NL[i]; ++l)
            fpWt[i][l] = prep(fpW[i][l], FP_K[i][l], FP_N[i][l]);
    _Float16* fcWt  = prep(fcW, 128, 128);
    _Float16* gdW1t = prep(gdW1, 128, 128);
    _Float16* asW1t = prep(asW1, 128, 128);
    _Float16* crW1t = prep(crW1, 135, 128);   // Kp=160
    _Float16* actW1t = prep(actW1, 138, 128); // Kp=160

    // ---- SA modules ----
    const int Ns[5]  = {4096, 1024, 256, 64, 16};
    const int NPs[4] = {1024, 256, 64, 16};
    const float RADs[4] = {0.1f, 0.2f, 0.4f, 0.8f};
    const int CFs[5] = {3, 64, 128, 256, 512};
    float* lxyz[5]  = {(float*)pcs, xyz1, xyz2, xyz3, xyz4};
    float* lfeat[5] = {(float*)pcs, f1, f2, f3, f4};

    for (int i = 0; i < 4; ++i) {
        int n = Ns[i], np = NPs[i], cf = CFs[i];
        int cinp = round32(3 + cf);
        int c1 = SA_N[i][0], c2 = SA_N[i][1], c3 = SA_N[i][2];

        fps_kernel<<<BATCH, 256, 0, stream>>>(lxyz[i], n, np, fpsidx);
        gather_xyz<<<ceil_div(BATCH * np, 256), 256, 0, stream>>>(
            lxyz[i], fpsidx, lxyz[i + 1], n, np);
        ball_query<<<ceil_div(BATCH * np, 128), 128, 0, stream>>>(
            lxyz[i], lxyz[i + 1], n, np, RADs[i] * RADs[i], gidx);

        for (int b = 0; b < BATCH; ++b) {
            const float* xb  = lxyz[i]     + (size_t)b * n  * 3;
            const float* nxb = lxyz[i + 1] + (size_t)b * np * 3;
            const float* fb  = lfeat[i]    + (size_t)b * n  * cf;
            const int*   gib = gidx        + (size_t)b * np * NSAMPLE;
            int rows = np * NSAMPLE;
            group_points_h<<<ceil_div(rows * cinp, 256), 256, 0, stream>>>(
                xb, nxb, fb, gib, bufA_h, np, cf, cinp);
            gemm(bufA_h, cinp, saWt[i][0], cinp, saB[i][0], c1,
                 bufB_h, c1, 1, 1, rows, cinp, 1, stream);
            gemm(bufB_h, c1, saWt[i][1], c1, saB[i][1], c2,
                 bufA_h, c2, 1, 1, rows, c1, 1, stream);
            gemm(bufA_h, c2, saWt[i][2], c2, saB[i][2], c3,
                 bufF, c3, 1, 0, rows, c2, 1, stream);
            maxpool32<<<ceil_div(np * c3, 256), 256, 0, stream>>>(
                bufF, lfeat[i + 1] + (size_t)b * np * c3, np, c3);
        }
    }

    // ---- FP modules (i = 3..0) ----
    for (int b = 0; b < BATCH; ++b) {  // i=3: 64 pts, K=768 -> 256 -> 256
        three_interp_h<<<1, 128, 0, stream>>>(
            xyz3 + (size_t)b*64*3, xyz4 + (size_t)b*16*3,
            f3 + (size_t)b*64*256, f4 + (size_t)b*16*512, bufA_h, 64, 16, 256, 512, 768);
        gemm(bufA_h, 768, fpWt[3][0], 768, fpB[3][0], 256, bufB_h, 256, 1, 1, 64, 768, 1, stream);
        gemm(bufB_h, 256, fpWt[3][1], 256, fpB[3][1], 256,
             nf3 + (size_t)b*64*256, 256, 1, 0, 64, 256, 1, stream);
    }
    for (int b = 0; b < BATCH; ++b) {  // i=2: 256 pts, K=384 -> 256 -> 256
        three_interp_h<<<2, 128, 0, stream>>>(
            xyz2 + (size_t)b*256*3, xyz3 + (size_t)b*64*3,
            f2 + (size_t)b*256*128, nf3 + (size_t)b*64*256, bufA_h, 256, 64, 128, 256, 384);
        gemm(bufA_h, 384, fpWt[2][0], 384, fpB[2][0], 256, bufB_h, 256, 1, 1, 256, 384, 1, stream);
        gemm(bufB_h, 256, fpWt[2][1], 256, fpB[2][1], 256,
             nf2 + (size_t)b*256*256, 256, 1, 0, 256, 256, 1, stream);
    }
    for (int b = 0; b < BATCH; ++b) {  // i=1: 1024 pts, K=320 -> 256 -> 128
        three_interp_h<<<8, 128, 0, stream>>>(
            xyz1 + (size_t)b*1024*3, xyz2 + (size_t)b*256*3,
            f1 + (size_t)b*1024*64, nf2 + (size_t)b*256*256, bufA_h, 1024, 256, 64, 256, 320);
        gemm(bufA_h, 320, fpWt[1][0], 320, fpB[1][0], 256, bufB_h, 256, 1, 1, 1024, 320, 1, stream);
        gemm(bufB_h, 256, fpWt[1][1], 256, fpB[1][1], 128,
             nf1 + (size_t)b*1024*128, 128, 1, 0, 1024, 256, 1, stream);
    }
    for (int b = 0; b < BATCH; ++b) {  // i=0: 4096 pts, K=131(pad 160) -> 128 -> 128 -> 128, then fc
        three_interp_h<<<32, 128, 0, stream>>>(
            (const float*)pcs + (size_t)b*4096*3, xyz1 + (size_t)b*1024*3,
            (const float*)pcs + (size_t)b*4096*3, nf1 + (size_t)b*1024*128,
            bufA_h, 4096, 1024, 3, 128, 160);
        gemm(bufA_h, 160, fpWt[0][0], 160, fpB[0][0], 128, bufB_h, 128, 1, 1, 4096, 160, 1, stream);
        gemm(bufB_h, 128, fpWt[0][1], 128, fpB[0][1], 128, bufA_h, 128, 1, 1, 4096, 128, 1, stream);
        gemm(bufA_h, 128, fpWt[0][2], 128, fpB[0][2], 128,
             f0fin + (size_t)b*4096*128, 128, 1, 1, 4096, 128, 1, stream);
        // fc + relu, stored transposed into whole[b] (channel-major)
        gemm(f0fin + (size_t)b*4096*128, 128, fcWt, 128, fcB, 128,
             whole + (size_t)b*128*4096, 1 /*rs*/, 4096 /*cs*/, 0, 4096, 128, 1, stream);
    }

    // ---- heads ----
    extract_net<<<ceil_div(BATCH * 128, 256), 256, 0, stream>>>(whole, net, net_h);

    // gdepth -> gd, width_loss
    gemm(net_h, 128, gdW1t, 128, gdB1, 128, h1, 128, 1, 0, 16, 128, 2, stream);
    dense_small<<<1, 32, 0, stream>>>(h1, gdW2, gdB2, gdv, 16, 128, 1, 0);
    width_loss_k<<<1, 32, 0, stream>>>(gdv, gtw, width_loss);

    // critic on [net | dirs1 dirs2 gd] -> logits, critic_loss
    build_crit1_h<<<ceil_div(16*160, 256), 256, 0, stream>>>(net, dirs1, dirs2, gdv, in_h);
    gemm(in_h, 160, crW1t, 160, crB1, 128, h1, 128, 1, 0, 16, 160, 2, stream);
    dense_small<<<1, 32, 0, stream>>>(h1, crW2, crB2, logits, 16, 128, 1, 0);
    critic_loss_k<<<1, 32, 0, stream>>>(logits, gtr, critic_loss);

    // actor -> pred6, theta, actor_cov
    build_actor_in_h<<<ceil_div(1600*160, 256), 256, 0, stream>>>(net, rvs, in_h);
    gemm(in_h, 160, actW1t, 160, actB1, 128, h1, 128, 1, 0, 1600, 160, 2, stream);
    dense_small<<<ceil_div(1600*6, 256), 256, 0, stream>>>(h1, actW2, actB2, o6, 1600, 128, 6, 0);
    actor_bgs<<<ceil_div(1600, 256), 256, 0, stream>>>(o6, pred6, 1600);
    theta_kernel<<<ceil_div(1600, 256), 256, 0, stream>>>(pred6, dirs1, dirs2, theta, 1600);
    min100_k<<<1, 32, 0, stream>>>(theta, actor_cov);

    // critic on [enet | pred6 | tiled gd] -> prop -> avg
    build_crit2_h<<<ceil_div(1600*160, 256), 256, 0, stream>>>(net, pred6, gdv, in_h);
    gemm(in_h, 160, crW1t, 160, crB1, 128, h1, 128, 1, 0, 1600, 160, 2, stream);
    dense_small<<<ceil_div(1600, 256), 256, 0, stream>>>(h1, crW2, crB2, prop, 1600, 128, 1, 3);
    mean100_k<<<1, 32, 0, stream>>>(prop, avg);

    // ascore -> pas, as_loss
    gemm(net_h, 128, asW1t, 128, asB1, 128, h1, 128, 1, 0, 16, 128, 2, stream);
    dense_small<<<1, 32, 0, stream>>>(h1, asW2, asB2, pas, 16, 128, 1, 3);
    as_loss_k<<<1, 32, 0, stream>>>(pas, avg, as_loss);
}
